// CenterGeoAttention_2886218023086
// MI455X (gfx1250) — compile-verified, matrix operand order checked by reading, not run on hardware
//
#include <hip/hip_runtime.h>
#include <hip/hip_bf16.h>

#define N_NODES 65536
#define D_DIM   1024
#define H_HEADS 16
#define HD_DIM  64
#define EPS     1e-5f
#define RES     0.5f

typedef __bf16 bf16x16 __attribute__((ext_vector_type(16)));
typedef __bf16 bf16x8  __attribute__((ext_vector_type(8)));
typedef float  floatx8 __attribute__((ext_vector_type(8)));

static __device__ __forceinline__ float fast_rcp(float x) {
#if __has_builtin(__builtin_amdgcn_rcpf)
  return __builtin_amdgcn_rcpf(x);
#else
  return 1.f / x;
#endif
}

// Concatenate two contiguous 8-element halves into a WMMA A/B fragment.
static __device__ __forceinline__ bf16x16 load_frag_rowmajor(const __bf16* p) {
  bf16x8 lo = *(const bf16x8*)p;
  bf16x8 hi = *(const bf16x8*)(p + 16);
  return __builtin_shufflevector(lo, hi, 0, 1, 2, 3, 4, 5, 6, 7,
                                 8, 9, 10, 11, 12, 13, 14, 15);
}

// ---------------------------------------------------------------------------
// K0: convert W1(top), Wg(top), W2 to bf16 in WMMA-fragment-swizzled layout:
//   block = nT*32 + kT  (nT: 16-col tile, kT: 32-k tile), 512 elems per block,
//   within block: lane*16 + j, so each lane's B fragment is 32 contiguous bytes.
// ---------------------------------------------------------------------------
__global__ void convert_weights_swz(const float* __restrict__ W1,
                                    const float* __restrict__ Wg,
                                    const float* __restrict__ W2,
                                    __bf16* __restrict__ W1b,
                                    __bf16* __restrict__ Wgb,
                                    __bf16* __restrict__ W2b) {
  size_t idx = (size_t)blockIdx.x * blockDim.x + threadIdx.x; // < 3*1M
  size_t which = idx >> 20;
  int o = (int)(idx & 0xFFFFF);
  int blk    = o >> 9;          // 2048 blocks = 64 nT * 32 kT
  int within = o & 511;
  int lane = within >> 4;
  int j    = within & 15;
  int nT = blk >> 5;
  int kT = blk & 31;
  int col = nT * 16 + (lane & 15);
  int k   = kT * 32 + ((lane >> 4) << 3) + (j & 7) + ((j >> 3) << 4);
  size_t src = (size_t)k * D_DIM + col;
  if (which == 0)      W1b[o] = (__bf16)W1[src];
  else if (which == 1) Wgb[o] = (__bf16)Wg[src];
  else                 W2b[o] = (__bf16)W2[src];
}

// ---------------------------------------------------------------------------
// K1: LN(h[c]) then q = h_c_ln @ Wq   (one block, 1024 threads)
// ---------------------------------------------------------------------------
__global__ void prep_center(const float* __restrict__ h, const int* __restrict__ cidx,
                            const float* __restrict__ gc, const float* __restrict__ bc,
                            const float* __restrict__ Wq, float* __restrict__ qv) {
  extern __shared__ char smem[];
  float* red  = (float*)smem;       // 1024
  float* red2 = red + 1024;         // 1024
  float* hcl  = red2 + 1024;        // 1024
  const int t = threadIdx.x;
  const int c = *cidx;
  float x = h[(size_t)c * D_DIM + t];
  red[t] = x; red2[t] = x * x;
  __syncthreads();
  for (int s = 512; s > 0; s >>= 1) {
    if (t < s) { red[t] += red[t + s]; red2[t] += red2[t + s]; }
    __syncthreads();
  }
  float mean = red[0] * (1.0f / D_DIM);
  float var  = red2[0] * (1.0f / D_DIM) - mean * mean;
  float rs   = rsqrtf(var + EPS);
  hcl[t] = (x - mean) * rs * gc[t] + bc[t];
  __syncthreads();
  float acc = 0.f;
  for (int e = 0; e < D_DIM; ++e) acc += hcl[e] * Wq[(size_t)e * D_DIM + t];
  qv[t] = acc;
}

// K2: Wkq[e][hh] = sum_d Wk[e, hh*64+d] * q[hh*64+d]
__global__ void fold_wkq(const float* __restrict__ Wk, const float* __restrict__ qv,
                         float* __restrict__ Wkq) {
  int idx = blockIdx.x * blockDim.x + threadIdx.x; // < 1024*16
  int e = idx >> 4, hh = idx & 15;
  float acc = 0.f;
  const float* wrow = Wk + (size_t)e * D_DIM + hh * HD_DIM;
  const float* qrow = qv + hh * HD_DIM;
  for (int d = 0; d < HD_DIM; ++d) acc += wrow[d] * qrow[d];
  Wkq[idx] = acc;
}

// K4: bias_logit[n][hh] = rbf[n]·Wb[0:64,hh] + sep[n]·Wb[64:96,hh]
__global__ void bias_logits(const float* __restrict__ rbf, const float* __restrict__ sep,
                            const float* __restrict__ Wb, float* __restrict__ blog) {
  size_t idx = (size_t)blockIdx.x * blockDim.x + threadIdx.x; // < N*16
  size_t n = idx >> 4; int hh = (int)(idx & 15);
  float acc = 0.f;
  const float* r = rbf + n * 64;
  for (int i = 0; i < 64; ++i) acc += r[i] * Wb[(size_t)i * 16 + hh];
  const float* s = sep + n * 32;
  for (int i = 0; i < 32; ++i) acc += s[i] * Wb[(size_t)(64 + i) * 16 + hh];
  blog[idx] = acc;
}

// K5: scattered local_bias contribution for the K=8192 neighbor rows
__global__ void scatter_bias(const int* __restrict__ nbr, const float* __restrict__ lb,
                             const float* __restrict__ Wb, float* __restrict__ blog) {
  int idx = blockIdx.x * blockDim.x + threadIdx.x; // < 8192*16
  int k = idx >> 4, hh = idx & 15;
  int n = nbr[k];
  float acc = 0.f;
  const float* l = lb + (size_t)k * 32;
  for (int i = 0; i < 32; ++i) acc += l[i] * Wb[(size_t)(96 + i) * 16 + hh];
  atomicAdd(&blog[(size_t)n * 16 + hh], acc);
}

// Kinit: zero A and S accumulators
__global__ void init_stats(float* __restrict__ A, float* __restrict__ S) {
  int t = threadIdx.x;
  for (int i = t; i < H_HEADS * D_DIM; i += 1024) A[i] = 0.f;
  if (t < H_HEADS) S[t] = 0.f;
}

// ---------------------------------------------------------------------------
// K6: per-row LN + logits[n,hh] = hln·Wkq/8 + bias ; stores (mean, rs) per row
// ---------------------------------------------------------------------------
__global__ void row_logits(const float* __restrict__ h,
                           const float* __restrict__ ga, const float* __restrict__ ba,
                           const float* __restrict__ Wkq, const float* __restrict__ blog,
                           float* __restrict__ logits, float* __restrict__ meanrs) {
  const int lane = threadIdx.x & 31;
  const int wid  = threadIdx.x >> 5;
  const size_t n = (size_t)blockIdx.x * 8 + wid;
  const size_t base = n * D_DIM;
  float s = 0.f, ss = 0.f;
  for (int e = lane; e < D_DIM; e += 32) {
    float x = h[base + e]; s += x; ss += x * x;
  }
  #pragma unroll
  for (int off = 16; off; off >>= 1) { s += __shfl_xor(s, off, 32); ss += __shfl_xor(ss, off, 32); }
  float mean = s * (1.0f / D_DIM);
  float rs   = rsqrtf(ss * (1.0f / D_DIM) - mean * mean + EPS);
  float acc[H_HEADS];
  #pragma unroll
  for (int hh = 0; hh < H_HEADS; ++hh) acc[hh] = 0.f;
  for (int e = lane; e < D_DIM; e += 32) {
    float x = (h[base + e] - mean) * rs * ga[e] + ba[e];
    const float* w = Wkq + (size_t)e * 16;
    #pragma unroll
    for (int hh = 0; hh < H_HEADS; ++hh) acc[hh] += x * w[hh];
  }
  #pragma unroll
  for (int hh = 0; hh < H_HEADS; ++hh) {
    #pragma unroll
    for (int off = 16; off; off >>= 1) acc[hh] += __shfl_xor(acc[hh], off, 32);
  }
  if (lane == 0) {
    meanrs[2 * n]     = mean;
    meanrs[2 * n + 1] = rs;
    #pragma unroll
    for (int hh = 0; hh < H_HEADS; ++hh)
      logits[n * 16 + hh] = acc[hh] * 0.125f + blog[n * 16 + hh];
  }
}

// K7: per-head max over all N rows (one block, 1024 threads)
__global__ void head_max(const float* __restrict__ logits, float* __restrict__ mh) {
  __shared__ float red[1024];
  const int t = threadIdx.x;
  const int hh = t & 15;
  float v = -1e30f;
  for (size_t i = (size_t)(t >> 4); i < N_NODES; i += 64)
    v = fmaxf(v, logits[i * 16 + hh]);
  red[t] = v; __syncthreads();
  for (int s = 512; s >= 16; s >>= 1) {
    if (t < s) red[t] = fmaxf(red[t], red[t + s]);
    __syncthreads();
  }
  if (t < 16) mh[t] = red[t];
}

// K8a: P[n,hh] = exp(logit - max); block-reduced S atomics (16 per block)
__global__ void compute_p(const float* __restrict__ logits, const float* __restrict__ mh,
                          float* __restrict__ P, float* __restrict__ S) {
  __shared__ float red[256];
  const int t = threadIdx.x;
  size_t idx = (size_t)blockIdx.x * 256 + t;
  int hh = (int)(idx & 15);
  float p = __expf(logits[idx] - mh[hh]);
  P[idx] = p;
  red[t] = p; __syncthreads();
  for (int s = 128; s >= 16; s >>= 1) {
    if (t < s) red[t] += red[t + s];
    __syncthreads();
  }
  if (t < 16) atomicAdd(&S[t], red[t]);
}

// ---------------------------------------------------------------------------
// K8b: A[hh,e] += sum_n P[n,hh] * hln[n,e]
//   64 blocks x 512 thr; each thread owns 2 e-cols x 16 heads in registers.
//   No per-element atomics: 16K global atomics per block at the end.
// ---------------------------------------------------------------------------
__global__ __launch_bounds__(512) void weighted_accum(
    const float* __restrict__ h, const float* __restrict__ ga,
    const float* __restrict__ ba, const float* __restrict__ meanrs,
    const float* __restrict__ P, float* __restrict__ A) {
  const int t = threadIdx.x;
  const int c0 = t * 2;
  float acc0[H_HEADS], acc1[H_HEADS];
  #pragma unroll
  for (int hh = 0; hh < H_HEADS; ++hh) { acc0[hh] = 0.f; acc1[hh] = 0.f; }
  const float g0 = ga[c0], g1 = ga[c0 + 1];
  const float be0 = ba[c0], be1 = ba[c0 + 1];
  const size_t n0 = (size_t)blockIdx.x * 1024;
  for (int r = 0; r < 1024; ++r) {
    const size_t n = n0 + r;
    float2 hv = *(const float2*)&h[n * D_DIM + c0];
    float mean = meanrs[2 * n], rs = meanrs[2 * n + 1];
    float x0 = (hv.x - mean) * rs * g0 + be0;
    float x1 = (hv.y - mean) * rs * g1 + be1;
    const float* pr = P + n * 16;   // thread-uniform -> scalar loads
    #pragma unroll
    for (int hh = 0; hh < H_HEADS; ++hh) {
      float p = pr[hh];
      acc0[hh] += p * x0;
      acc1[hh] += p * x1;
    }
  }
  #pragma unroll
  for (int hh = 0; hh < H_HEADS; ++hh) {
    atomicAdd(&A[hh * D_DIM + c0],     acc0[hh]);
    atomicAdd(&A[hh * D_DIM + c0 + 1], acc1[hh]);
  }
}

// ---------------------------------------------------------------------------
// K9: out_center -> h_c_new -> c1/cg constant vectors  (one block, 1024 thr)
// ---------------------------------------------------------------------------
__global__ void center_consts(const float* __restrict__ h, const int* __restrict__ cidx,
                              const float* __restrict__ A, const float* __restrict__ S,
                              const float* __restrict__ Wv, const float* __restrict__ Wo,
                              const float* __restrict__ W1, const float* __restrict__ b1,
                              const float* __restrict__ Wg, const float* __restrict__ bg,
                              float* __restrict__ hcn_out, float* __restrict__ c1,
                              float* __restrict__ cg) {
  extern __shared__ char smem[];
  float* wbar = (float*)smem;            // 16*1024
  float* oc   = wbar + H_HEADS * D_DIM;  // 1024
  float* hcl  = oc + D_DIM;              // 1024
  const int t = threadIdx.x;
  for (int i = t; i < H_HEADS * D_DIM; i += 1024) wbar[i] = A[i] / S[i >> 10];
  __syncthreads();
  const int hh = t >> 6, d = t & 63;
  float acc = 0.f;
  for (int e = 0; e < D_DIM; ++e)
    acc += wbar[hh * D_DIM + e] * Wv[(size_t)e * D_DIM + hh * HD_DIM + d];
  oc[t] = acc;                      // flattened (H,HD) == D
  __syncthreads();
  const int c = *cidx;
  float acc2 = 0.f;
  for (int i = 0; i < D_DIM; ++i) acc2 += oc[i] * Wo[(size_t)i * D_DIM + t];
  float hcnv = h[(size_t)c * D_DIM + t] + RES * acc2;
  hcl[t] = hcnv; hcn_out[t] = hcnv;
  __syncthreads();
  float a1 = b1[t], ag = bg[t];
  for (int e = 0; e < D_DIM; ++e) {
    float x = hcl[e];
    a1 += x * W1[(size_t)(D_DIM + e) * D_DIM + t];
    ag += x * Wg[(size_t)(D_DIM + e) * D_DIM + t];
  }
  c1[t] = a1; cg[t] = ag;
}

// ---------------------------------------------------------------------------
// K10: fused gated MLP — the heavy WMMA kernel.
//   16 node rows per block; 8 waves x 16 cols/iter; swizzled bf16 weights so
//   each B fragment is one contiguous 32B load per lane.
// ---------------------------------------------------------------------------
__global__ __launch_bounds__(256) void mlp_wmma(
    const float* __restrict__ h,
    const __bf16* __restrict__ W1b, const __bf16* __restrict__ Wgb,
    const __bf16* __restrict__ W2b,
    const float* __restrict__ c1, const float* __restrict__ cg,
    const float* __restrict__ b2, float* __restrict__ out) {
  extern __shared__ char smem[];
  __bf16* Xs = (__bf16*)smem;      // 16*1024 bf16
  __bf16* Us = Xs + 16 * D_DIM;    // 16*1024 bf16
  __bf16* Gs = Us + 16 * D_DIM;    // 16*1024 bf16
  const int tid  = threadIdx.x;
  const int lane = tid & 31;
  const int wid  = tid >> 5;
  const int n0   = blockIdx.x * 16;

  // X tile load: float4 global reads, bf16 convert
  for (int idx = tid; idx < 16 * 256; idx += 256) {
    int r = idx >> 8, q = idx & 255;
    const float4 v = *(const float4*)&h[(size_t)(n0 + r) * D_DIM + q * 4];
    __bf16* d = Xs + r * D_DIM + q * 4;
    d[0] = (__bf16)v.x; d[1] = (__bf16)v.y; d[2] = (__bf16)v.z; d[3] = (__bf16)v.w;
  }
  __syncthreads();

  const int row  = lane & 15;           // A-frag row (M)
  const int koff = (lane >> 4) << 3;    // lanes 16-31 carry K+8 / K+24 halves
  const int lofs = lane << 4;           // lane offset inside swizzled block

  // ---- stage 1: U = silu(X@W1 + c1), G = sigmoid(X@Wg + cg) ----
  for (int it = 0; it < 8; ++it) {
    const int cb  = it * 128 + wid * 16;
    const int nT  = cb >> 4;
    const int col = cb + (lane & 15);
    floatx8 aU = {}; floatx8 aG = {};
    for (int kk = 0; kk < D_DIM; kk += 32) {
      const size_t blkoff = ((size_t)(nT * 32 + (kk >> 5)) << 9) + lofs;
      __builtin_prefetch(&W1b[blkoff + 1024], 0, 0);
      bf16x16 af  = load_frag_rowmajor(Xs + row * D_DIM + kk + koff);
      bf16x16 b1f = *(const bf16x16*)(W1b + blkoff);
      bf16x16 bgf = *(const bf16x16*)(Wgb + blkoff);
      aU = __builtin_amdgcn_wmma_f32_16x16x32_bf16(false, af, false, b1f, (short)0, aU, false, false);
      aG = __builtin_amdgcn_wmma_f32_16x16x32_bf16(false, af, false, bgf, (short)0, aG, false, false);
    }
    const float c1v = c1[col], cgv = cg[col];
    #pragma unroll
    for (int r2 = 0; r2 < 8; ++r2) {
      int m = ((lane >> 4) << 3) + r2;
      float u  = aU[r2] + c1v;
      float su = u * fast_rcp(1.f + __expf(-u));
      float g  = fast_rcp(1.f + __expf(-(aG[r2] + cgv)));
      Us[m * D_DIM + col] = (__bf16)su;
      Gs[m * D_DIM + col] = (__bf16)g;
    }
  }
  __syncthreads();

  // ---- stage 2: F = U@W2 + b2 ; out = h + 0.5*G*F ----
  for (int it = 0; it < 8; ++it) {
    const int cb  = it * 128 + wid * 16;
    const int nT  = cb >> 4;
    const int col = cb + (lane & 15);
    floatx8 aF = {};
    for (int kk = 0; kk < D_DIM; kk += 32) {
      const size_t blkoff = ((size_t)(nT * 32 + (kk >> 5)) << 9) + lofs;
      bf16x16 af  = load_frag_rowmajor(Us + row * D_DIM + kk + koff);
      bf16x16 b2f = *(const bf16x16*)(W2b + blkoff);
      aF = __builtin_amdgcn_wmma_f32_16x16x32_bf16(false, af, false, b2f, (short)0, aF, false, false);
    }
    const float b2v = b2[col];
    #pragma unroll
    for (int r2 = 0; r2 < 8; ++r2) {
      int m = ((lane >> 4) << 3) + r2;
      float f = aF[r2] + b2v;
      float g = (float)Gs[m * D_DIM + col];
      size_t oidx = (size_t)(n0 + m) * D_DIM + col;
      out[oidx] = h[oidx] + RES * g * f;
    }
  }
}

// K11: overwrite center row with h_c_new
__global__ void fix_center(const int* __restrict__ cidx, const float* __restrict__ hcn,
                           float* __restrict__ out) {
  const int c = *cidx;
  out[(size_t)c * D_DIM + threadIdx.x] = hcn[threadIdx.x];
}

// ---------------------------------------------------------------------------
extern "C" void kernel_launch(void* const* d_in, const int* in_sizes, int n_in,
                              void* d_out, int out_size, void* d_ws, size_t ws_size,
                              hipStream_t stream) {
  const float* h    = (const float*)d_in[0];
  const int*   cidx = (const int*)d_in[1];
  const float* rbf  = (const float*)d_in[2];
  const float* sep  = (const float*)d_in[3];
  const int*   nbr  = (const int*)d_in[4];
  const float* lb   = (const float*)d_in[5];
  const float* gc   = (const float*)d_in[6];
  const float* bc   = (const float*)d_in[7];
  const float* ga   = (const float*)d_in[8];
  const float* ba   = (const float*)d_in[9];
  const float* Wq   = (const float*)d_in[10];
  const float* Wk   = (const float*)d_in[11];
  const float* Wv   = (const float*)d_in[12];
  const float* Wo   = (const float*)d_in[13];
  const float* Wb   = (const float*)d_in[14];
  const float* W1   = (const float*)d_in[15];
  const float* b1   = (const float*)d_in[16];
  const float* W2   = (const float*)d_in[17];
  const float* b2   = (const float*)d_in[18];
  const float* Wg   = (const float*)d_in[19];
  const float* bg   = (const float*)d_in[20];
  float* out = (float*)d_out;

  // ---- workspace carve (~20 MB total) ----
  char* ws = (char*)d_ws;
  __bf16* W1b = (__bf16*)ws;
  __bf16* Wgb = W1b + (1u << 20);
  __bf16* W2b = Wgb + (1u << 20);
  float* fws    = (float*)(ws + 3ull * (1u << 20) * sizeof(__bf16));
  float* Wkq    = fws;                                 // 1024*16
  float* blog   = Wkq + 1024 * 16;                     // N*16
  float* logits = blog + (size_t)N_NODES * 16;         // N*16
  float* P      = logits + (size_t)N_NODES * 16;       // N*16
  float* meanrs = P + (size_t)N_NODES * 16;            // 2*N
  float* A      = meanrs + (size_t)2 * N_NODES;        // 16*1024
  float* S      = A + 16 * 1024;                       // 16
  float* mh     = S + 16;                              // 16
  float* qv     = mh + 16;                             // 1024
  float* hcn    = qv + 1024;                           // 1024
  float* c1     = hcn + 1024;                          // 1024
  float* cg     = c1 + 1024;                           // 1024

  convert_weights_swz<<<3 * 1024, 1024, 0, stream>>>(W1, Wg, W2, W1b, Wgb, W2b);
  prep_center<<<1, 1024, 3 * 1024 * sizeof(float), stream>>>(h, cidx, gc, bc, Wq, qv);
  fold_wkq<<<64, 256, 0, stream>>>(Wk, qv, Wkq);
  bias_logits<<<(N_NODES * 16) / 256, 256, 0, stream>>>(rbf, sep, Wb, blog);
  scatter_bias<<<(8192 * 16) / 256, 256, 0, stream>>>(nbr, lb, Wb, blog);
  init_stats<<<1, 1024, 0, stream>>>(A, S);
  row_logits<<<N_NODES / 8, 256, 0, stream>>>(h, ga, ba, Wkq, blog, logits, meanrs);
  head_max<<<1, 1024, 0, stream>>>(logits, mh);
  compute_p<<<(N_NODES * 16) / 256, 256, 0, stream>>>(logits, mh, P, S);
  weighted_accum<<<64, 512, 0, stream>>>(h, ga, ba, meanrs, P, A);
  center_consts<<<1, 1024, (16 * 1024 + 2 * 1024) * sizeof(float), stream>>>(
      h, cidx, A, S, Wv, Wo, W1, b1, Wg, bg, hcn, c1, cg);
  mlp_wmma<<<N_NODES / 16, 256, 3 * 16 * 1024 * sizeof(__bf16), stream>>>(
      h, W1b, Wgb, W2b, c1, cg, b2, out);
  fix_center<<<1, 1024, 0, stream>>>(cidx, hcn, out);
}
